// APNBBlock_85736137162914
// MI455X (gfx1250) — matrix operand order repaired and checked by reading, not compiled, not run on hardware
//
#include <hip/hip_runtime.h>

typedef __attribute__((ext_vector_type(16))) __bf16 v16bf;
typedef __attribute__((ext_vector_type(8)))  float  v8f;

// CDNA5 async global->LDS path (ASYNCcnt), guarded so missing builtins (or the
// host-side compile pass) fall back to register staging.
// Probe round 3: param type is 'vector_size(16) int __device__ *' (addrspace 1).
#if __has_builtin(__builtin_amdgcn_global_load_async_to_lds_b128) && \
    __has_builtin(__builtin_amdgcn_s_wait_asynccnt)
#define HAS_ASYNC_LDS 1
typedef int b128_i __attribute__((vector_size(16)));
typedef __attribute__((address_space(1))) b128_i* gbl_b128_p;
typedef __attribute__((address_space(3))) b128_i* lds_b128_p;
#else
#define HAS_ASYNC_LDS 0
#endif

constexpr int kHW     = 14400;   // 120*120
constexpr int kCin    = 512;
constexpr int kCk     = 256;
constexpr int kCv     = 256;
constexpr int kCout   = 512;
constexpr int kBatch  = 4;
constexpr int kSValid = 110;     // 1+9+36+64
constexpr int kSPad   = 128;
constexpr int kSTiles = 7;       // 112 computed cols; 110,111 masked

__device__ __forceinline__ unsigned short f2bf(float f) {
  union { float f; unsigned u; } v; v.f = f;
  unsigned r = v.u + 0x7FFFu + ((v.u >> 16) & 1u);
  return (unsigned short)(r >> 16);
}
__device__ __forceinline__ float bf2f(unsigned short h) {
  union { unsigned u; float f; } v; v.u = ((unsigned)h) << 16;
  return v.f;
}

// ---------------------------------------------------------------------------
// K0: fold BN into Wk, pack [Wk';Wv] -> bf16 [512][512], Wo -> bf16, bias'
// ---------------------------------------------------------------------------
__global__ void prep_kernel(const float* __restrict__ Wk, const float* __restrict__ bk,
                            const float* __restrict__ gamma, const float* __restrict__ beta,
                            const float* __restrict__ mean, const float* __restrict__ var,
                            const float* __restrict__ Wv, const float* __restrict__ bv,
                            const float* __restrict__ Wo,
                            unsigned short* __restrict__ Wkv_bf,
                            unsigned short* __restrict__ Wo_bf,
                            float* __restrict__ bias) {
  int i = blockIdx.x * blockDim.x + threadIdx.x;
  const int NW1 = kCk * kCin, NW2 = kCv * kCin, NWO = kCout * kCv;
  if (i < NW1) {
    int co = i / kCin;
    float inv = gamma[co] * rsqrtf(var[co] + 1e-5f);
    Wkv_bf[i] = f2bf(Wk[i] * inv);
  } else if (i < NW1 + NW2) {
    Wkv_bf[i] = f2bf(Wv[i - NW1]);
  } else if (i < NW1 + NW2 + NWO) {
    int j = i - NW1 - NW2;
    Wo_bf[j] = f2bf(Wo[j]);
  } else if (i < NW1 + NW2 + NWO + kCin) {
    int co = i - NW1 - NW2 - NWO;
    if (co < kCk) {
      float inv = gamma[co] * rsqrtf(var[co] + 1e-5f);
      bias[co] = bk[co] * inv + beta[co] - mean[co] * inv;
    } else {
      bias[co] = bv[co - kCk];
    }
  }
}

// ---------------------------------------------------------------------------
// WMMA GEMM: C[b][m][n] = A[m][k] * B[b][k][n] + bias[m], optional ReLU rows,
// optional transposed bf16 copy q_t[b][n][m] for m<256.
// Block = 256 threads (8 waves), tile 128x128x32, per-wave 32x64 via 2x4 WMMAs.
// Double-buffered LDS pipeline, one barrier per K-step. A-tile staged via
// GLOBAL_LOAD_ASYNC_TO_LDS_B128 (ASYNCcnt) when available; B-tile staged via
// registers (needs transpose + f32->bf16). global_prefetch warms tile k+2.
// ---------------------------------------------------------------------------
template <int KDIM, bool IN_BF16, bool OUT_F32, bool WRITE_QT>
__launch_bounds__(256)
__global__ void gemm_wmma_kernel(const unsigned short* __restrict__ A,
                                 const void* __restrict__ Bsrc,
                                 const float* __restrict__ bias,
                                 void* __restrict__ Cdst,
                                 unsigned short* __restrict__ q_t,
                                 int reluRows) {
  // pad rows to 40 shorts (80B): 16-lane frag reads hit 16 distinct bank quads
  __shared__ __align__(16) unsigned short sA[2][128][40];
  __shared__ __align__(16) unsigned short sB[2][128][40];   // transposed: sB[n][k]

  const int b  = blockIdx.z;
  const int m0 = blockIdx.y * 128;
  const int n0 = blockIdx.x * 128;
  const int t  = threadIdx.x;
  const int lane = t & 31;
  const int half = lane >> 4;
  const int l    = lane & 15;
  const int wm   = (t >> 5) & 3;   // M sub-tile (32 rows)
  const int wn   = (t >> 5) >> 2;  // N sub-tile (64 cols)
  const bool fullN = (n0 + 128 <= kHW);

  v8f acc[2][4];
#pragma unroll
  for (int mt = 0; mt < 2; ++mt)
#pragma unroll
    for (int nt = 0; nt < 4; ++nt)
#pragma unroll
      for (int i = 0; i < 8; ++i) acc[mt][nt][i] = 0.0f;

  // staging registers for the in-flight tile
  uint4  stA[2];
  uint4  stBh[2];
  float4 stBf[4];
  (void)stA;

  // issue tile-k0 loads: A goes straight to LDS buf (async) or to registers
  auto fetch = [&](int k0, int buf) {
    (void)buf;
#pragma unroll
    for (int p = 0; p < 2; ++p) {
      int idx = p * 256 + t, row = idx >> 2, ko = (idx & 3) * 8;
#if HAS_ASYNC_LDS
      __builtin_amdgcn_global_load_async_to_lds_b128(
          (gbl_b128_p)(A + (size_t)(m0 + row) * KDIM + (k0 + ko)),
          (lds_b128_p)&sA[buf][row][ko], 0, 0);
#else
      stA[p] = *(const uint4*)(A + (size_t)(m0 + row) * KDIM + (k0 + ko));
#endif
    }
    if constexpr (IN_BF16) {
      const unsigned short* Bp = (const unsigned short*)Bsrc;
#pragma unroll
      for (int p = 0; p < 2; ++p) {
        int idx = p * 256 + t, kr = idx >> 4, nc = (idx & 15) * 8;
        const unsigned short* src =
            Bp + ((size_t)b * KDIM + (k0 + kr)) * kHW + (n0 + nc);
        if (fullN) {
          stBh[p] = *(const uint4*)src;
        } else {
          union { uint4 q; unsigned short s[8]; } vv;
#pragma unroll
          for (int i = 0; i < 8; ++i)
            vv.s[i] = (n0 + nc + i < kHW) ? src[i] : (unsigned short)0;
          stBh[p] = vv.q;
        }
      }
    } else {
      const float* Bp = (const float*)Bsrc;
#pragma unroll
      for (int p = 0; p < 4; ++p) {
        int idx = p * 256 + t, kr = idx >> 5, nc = (idx & 31) * 4;
        const float* src = Bp + ((size_t)b * KDIM + (k0 + kr)) * kHW + (n0 + nc);
        if (fullN) {
          stBf[p] = *(const float4*)src;
        } else {
          float4 f;
          f.x = (n0 + nc + 0 < kHW) ? src[0] : 0.0f;
          f.y = (n0 + nc + 1 < kHW) ? src[1] : 0.0f;
          f.z = (n0 + nc + 2 < kHW) ? src[2] : 0.0f;
          f.w = (n0 + nc + 3 < kHW) ? src[3] : 0.0f;
          stBf[p] = f;
        }
      }
    }
  };

  // finish tile staging into LDS buffer `buf`
  auto commit = [&](int buf) {
#if !HAS_ASYNC_LDS
#pragma unroll
    for (int p = 0; p < 2; ++p) {
      int idx = p * 256 + t, row = idx >> 2, ko = (idx & 3) * 8;
      *(uint4*)&sA[buf][row][ko] = stA[p];
    }
#endif
    if constexpr (IN_BF16) {
#pragma unroll
      for (int p = 0; p < 2; ++p) {
        int idx = p * 256 + t, kr = idx >> 4, nc = (idx & 15) * 8;
        union { uint4 q; unsigned short s[8]; } vv; vv.q = stBh[p];
#pragma unroll
        for (int i = 0; i < 8; ++i) sB[buf][nc + i][kr] = vv.s[i];
      }
    } else {
#pragma unroll
      for (int p = 0; p < 4; ++p) {
        int idx = p * 256 + t, kr = idx >> 5, nc = (idx & 31) * 4;
        sB[buf][nc + 0][kr] = f2bf(stBf[p].x);
        sB[buf][nc + 1][kr] = f2bf(stBf[p].y);
        sB[buf][nc + 2][kr] = f2bf(stBf[p].z);
        sB[buf][nc + 3][kr] = f2bf(stBf[p].w);
      }
    }
#if HAS_ASYNC_LDS
    __builtin_amdgcn_s_wait_asynccnt(0);   // A-tile DMA complete before barrier
#endif
  };

  fetch(0, 0);
  commit(0);
  __syncthreads();
  int cur = 0;

  for (int k0 = 0; k0 < KDIM; k0 += 32) {
    const bool hasNext = (k0 + 32 < KDIM);
    if (hasNext) {
      fetch(k0 + 32, 1 - cur);              // loads overlap the WMMAs below
      if (k0 + 64 < KDIM) {                 // warm tile k+2 (global_prefetch_b8)
        __builtin_prefetch(A + (size_t)(m0 + (t >> 2)) * KDIM + (k0 + 64), 0, 1);
        if constexpr (IN_BF16)
          __builtin_prefetch((const unsigned short*)Bsrc +
                                 ((size_t)b * KDIM + (k0 + 64 + (t >> 4))) * kHW + n0,
                             0, 1);
        else
          __builtin_prefetch((const float*)Bsrc +
                                 ((size_t)b * KDIM + (k0 + 64 + (t >> 5))) * kHW + n0,
                             0, 1);
      }
    }

    // ---- fragments + 2x4 WMMAs on buffer `cur` ----
    union AF { v16bf v; uint4 q[2]; };
    AF a[2], bfr;
#pragma unroll
    for (int mt = 0; mt < 2; ++mt) {
      int r = wm * 32 + mt * 16 + l;
      // A 16x32 bf16 layout: lanes 0-15 K{0..7,16..23}; lanes 16-31 K{8..15,24..31}
      a[mt].q[0] = *(const uint4*)&sA[cur][r][half * 8];
      a[mt].q[1] = *(const uint4*)&sA[cur][r][16 + half * 8];
    }
#pragma unroll
    for (int nt = 0; nt < 4; ++nt) {
      int n = wn * 64 + nt * 16 + l;
      // B 32x16 bf16 layout: lane = N, lanes 0-15 K{0..7,16..23}; lanes 16-31 K{8..15,24..31}
      bfr.q[0] = *(const uint4*)&sB[cur][n][half * 8];
      bfr.q[1] = *(const uint4*)&sB[cur][n][16 + half * 8];
#pragma unroll
      for (int mt = 0; mt < 2; ++mt) {
        acc[mt][nt] = __builtin_amdgcn_wmma_f32_16x16x32_bf16(
            false, a[mt].v, false, bfr.v, (short)0, acc[mt][nt], false, false);
      }
    }

    if (hasNext) {
      commit(1 - cur);    // safe: buf (1-cur) was last read before prior barrier
      __syncthreads();
      cur ^= 1;
    }
  }

  // ---- epilogue: bias (+ReLU rows), store bf16 or f32, optional q_t copy ----
#pragma unroll
  for (int mt = 0; mt < 2; ++mt) {
#pragma unroll
    for (int nt = 0; nt < 4; ++nt) {
#pragma unroll
      for (int v = 0; v < 8; ++v) {
        int mg = m0 + wm * 32 + mt * 16 + half * 8 + v;  // C: vgpr = M, lane = N
        int ng = n0 + wn * 64 + nt * 16 + l;
        if (ng < kHW) {
          float val = acc[mt][nt][v] + bias[mg];
          if (mg < reluRows) val = fmaxf(val, 0.0f);
          size_t off = ((size_t)b * kCout + mg) * kHW + ng;
          if constexpr (OUT_F32) ((float*)Cdst)[off] = val;
          else ((unsigned short*)Cdst)[off] = f2bf(val);
          if constexpr (WRITE_QT) {
            if (mg < kCk)
              q_t[((size_t)b * kHW + ng) * kCk + mg] = f2bf(val);
          }
        }
      }
    }
  }
}

// ---------------------------------------------------------------------------
// K2: pyramid pooling via shared 24x24 grid of 5x5 base sums (120 = lcm tiling
// for p=1,3,6,8). Emits key_t[b][s][c] (c-contig) and val[b][c][s] (s-contig),
// zero-padded to S=128, so attention B-fragments are contiguous 16B loads.
// ---------------------------------------------------------------------------
__global__ void ppm_kernel(const unsigned short* __restrict__ kqv,
                           unsigned short* __restrict__ key_t,
                           unsigned short* __restrict__ val) {
  const int c512 = blockIdx.x;   // 0..255 -> key (from kq), 256..511 -> value (from v)
  const int b = blockIdx.y;
  const int t = threadIdx.x;     // 256 threads
  __shared__ float base[576];
  const unsigned short* src = kqv + ((size_t)b * 512 + c512) * kHW;
  for (int cell = t; cell < 576; cell += 256) {
    int ci = cell / 24, cj = cell % 24;
    float s = 0.0f;
    for (int r = 0; r < 5; ++r)
      for (int cc = 0; cc < 5; ++cc)
        s += bf2f(src[(ci * 5 + r) * 120 + cj * 5 + cc]);
    base[cell] = s;
  }
  __syncthreads();
  if (t < kSPad) {
    float out = 0.0f;
    if (t < kSValid) {
      int p, idx;
      if (t < 1)       { p = 1; idx = 0; }
      else if (t < 10) { p = 3; idx = t - 1; }
      else if (t < 46) { p = 6; idx = t - 10; }
      else             { p = 8; idx = t - 46; }
      int pi = idx / p, pj = idx % p;
      int bc = 24 / p;                      // base cells per pooled cell side
      float s = 0.0f;
      for (int r = 0; r < bc; ++r)
        for (int cc = 0; cc < bc; ++cc)
          s += base[(pi * bc + r) * 24 + pj * bc + cc];
      float cnt = (120.0f / p) * (120.0f / p);
      out = s / cnt;
    }
    unsigned short ob = f2bf(out);
    if (c512 < kCk) key_t[((size_t)b * kSPad + t) * kCk + c512] = ob;
    else            val[((size_t)b * kCv + (c512 - kCk)) * kSPad + t] = ob;
  }
}

// ---------------------------------------------------------------------------
// K3: fused scores -> softmax -> agg. Wave handles 16 query rows.
// scores: 7 S-tiles x (K=256 in 8 WMMA steps); softmax in registers with
// width-16 shfl_xor reductions; probs re-laid C->A through 4KB/wave LDS;
// agg: 16 N-tiles x (K=128 in 4 WMMA steps) against L2-resident value.
// ---------------------------------------------------------------------------
__launch_bounds__(256)
__global__ void attn_kernel(const unsigned short* __restrict__ q_t,
                            const unsigned short* __restrict__ key_t,
                            const unsigned short* __restrict__ val,
                            unsigned short* __restrict__ agg) {
  const int b   = blockIdx.y;
  const int hw0 = blockIdx.x * 128;
  const int t = threadIdx.x;
  const int w = t >> 5;
  const int lane = t & 31;
  const int half = lane >> 4;
  const int l    = lane & 15;

  __shared__ __align__(16) unsigned short attn_lds[8][16][kSPad];  // 32KB

  union AF { v16bf v; uint4 q[2]; };
  AF a, bfr;

  // ---- scores: 16 x 112 (padded), K = 256 ----
  int mrow = hw0 + w * 16 + l;
  const unsigned short* qrow =
      q_t + ((size_t)b * kHW + (mrow < kHW ? mrow : kHW - 1)) * kCk;

  v8f sc[kSTiles];
#pragma unroll
  for (int st = 0; st < kSTiles; ++st)
#pragma unroll
    for (int i = 0; i < 8; ++i) sc[st][i] = 0.0f;

  for (int k0 = 0; k0 < kCk; k0 += 32) {
    a.q[0] = *(const uint4*)(qrow + k0 + half * 8);
    a.q[1] = *(const uint4*)(qrow + k0 + 16 + half * 8);
#pragma unroll
    for (int st = 0; st < kSTiles; ++st) {
      const unsigned short* krow =
          key_t + ((size_t)b * kSPad + st * 16 + l) * kCk + k0 + half * 8;
      bfr.q[0] = *(const uint4*)(krow);
      bfr.q[1] = *(const uint4*)(krow + 16);
      sc[st] = __builtin_amdgcn_wmma_f32_16x16x32_bf16(
          false, a.v, false, bfr.v, (short)0, sc[st], false, false);
    }
  }

  // ---- softmax over S (cols live in lanes 0..15 of each half, 7 tiles) ----
  const float scale = 0.0625f;  // 1/sqrt(256)
#pragma unroll
  for (int v = 0; v < 8; ++v) {
    float m = -3.4e38f;
#pragma unroll
    for (int st = 0; st < kSTiles; ++st) {
      float s = sc[st][v] * scale;
      if (st == kSTiles - 1 && l >= (kSValid - (kSTiles - 1) * 16)) s = -3.4e38f;
      sc[st][v] = s;
      m = fmaxf(m, s);
    }
    m = fmaxf(m, __shfl_xor(m, 1, 16));
    m = fmaxf(m, __shfl_xor(m, 2, 16));
    m = fmaxf(m, __shfl_xor(m, 4, 16));
    m = fmaxf(m, __shfl_xor(m, 8, 16));
    float sum = 0.0f;
#pragma unroll
    for (int st = 0; st < kSTiles; ++st) {
      float e = __expf(sc[st][v] - m);
      sc[st][v] = e;
      sum += e;
    }
    sum += __shfl_xor(sum, 1, 16);
    sum += __shfl_xor(sum, 2, 16);
    sum += __shfl_xor(sum, 4, 16);
    sum += __shfl_xor(sum, 8, 16);
    float inv = 1.0f / sum;
    int r = half * 8 + v;   // local query row 0..15
#pragma unroll
    for (int st = 0; st < kSTiles; ++st)
      attn_lds[w][r][st * 16 + l] = f2bf(sc[st][v] * inv);
    attn_lds[w][r][112 + l] = 0;   // zero pad cols 112..127
  }

  // ---- agg: 16 x 256, K = S_pad = 128 ----
  const unsigned short* at = &attn_lds[w][0][0];
#pragma unroll
  for (int nt = 0; nt < 16; ++nt) {
    v8f accv;
#pragma unroll
    for (int i = 0; i < 8; ++i) accv[i] = 0.0f;
#pragma unroll
    for (int kt = 0; kt < 4; ++kt) {
      a.q[0] = *(const uint4*)(at + l * kSPad + kt * 32 + half * 8);
      a.q[1] = *(const uint4*)(at + l * kSPad + kt * 32 + 16 + half * 8);
      const unsigned short* vrow =
          val + ((size_t)b * kCv + nt * 16 + l) * kSPad + kt * 32 + half * 8;
      bfr.q[0] = *(const uint4*)(vrow);
      bfr.q[1] = *(const uint4*)(vrow + 16);
      accv = __builtin_amdgcn_wmma_f32_16x16x32_bf16(
          false, a.v, false, bfr.v, (short)0, accv, false, false);
    }
#pragma unroll
    for (int v = 0; v < 8; ++v) {
      int m = hw0 + w * 16 + half * 8 + v;
      int c = nt * 16 + l;
      if (m < kHW)
        agg[((size_t)b * kCv + c) * kHW + m] = f2bf(accv[v]);
    }
  }
}

// ---------------------------------------------------------------------------
extern "C" void kernel_launch(void* const* d_in, const int* in_sizes, int n_in,
                              void* d_out, int out_size, void* d_ws, size_t ws_size,
                              hipStream_t stream) {
  (void)in_sizes; (void)n_in; (void)out_size; (void)ws_size;
  const float* x     = (const float*)d_in[0];
  const float* Wk    = (const float*)d_in[1];
  const float* bk    = (const float*)d_in[2];
  const float* gamma = (const float*)d_in[3];
  const float* beta  = (const float*)d_in[4];
  const float* mean  = (const float*)d_in[5];
  const float* var   = (const float*)d_in[6];
  const float* Wv    = (const float*)d_in[7];
  const float* bv    = (const float*)d_in[8];
  const float* Wo    = (const float*)d_in[9];
  const float* bo    = (const float*)d_in[10];

  char* ws = (char*)d_ws;
  unsigned short* Wkv_bf = (unsigned short*)(ws + 0);           // 512*512*2
  unsigned short* Wo_bf  = (unsigned short*)(ws + 524288);      // 512*256*2
  float*          bias   = (float*)(ws + 786432);               // 512*4
  unsigned short* kqv    = (unsigned short*)(ws + 788480);      // 4*512*14400*2
  unsigned short* q_t    = (unsigned short*)(ws + 59770880);    // 4*14400*256*2
  unsigned short* key_t  = (unsigned short*)(ws + 89262080);    // 4*128*256*2
  unsigned short* val    = (unsigned short*)(ws + 89524224);    // 4*256*128*2
  unsigned short* agg    = (unsigned short*)(ws + 89786368);    // 4*256*14400*2

  {
    int total = 512 * 512 + 512 * 256 + 512;
    prep_kernel<<<(total + 255) / 256, 256, 0, stream>>>(
        Wk, bk, gamma, beta, mean, var, Wv, bv, Wo, Wkv_bf, Wo_bf, bias);
  }

  dim3 gGemm((kHW + 127) / 128, kCout / 128, kBatch);
  // conv_kq + conv_v fused (BN folded, ReLU on rows<256), writes kqv + q_t
  gemm_wmma_kernel<kCin, false, false, true>
      <<<gGemm, 256, 0, stream>>>(Wkv_bf, x, bias, kqv, q_t, kCk);

  ppm_kernel<<<dim3(512, kBatch), 256, 0, stream>>>(kqv, key_t, val);

  attn_kernel<<<dim3((kHW + 127) / 128, kBatch), 256, 0, stream>>>(
      q_t, key_t, val, agg);

  // conv_out: Wo @ agg + bo -> f32 output
  gemm_wmma_kernel<kCv, true, true, false>
      <<<gGemm, 256, 0, stream>>>(Wo_bf, agg, bo, d_out, nullptr, 0);
}
